// CNF_52828097740860
// MI455X (gfx1250) — matrix operand order repaired and checked by reading, not compiled, not run on hardware
//
#include <hip/hip_runtime.h>

// ---------------------------------------------------------------------------
// CNF exact-trace RK4 integrator for MI455X (gfx1250, wave32, WMMA).
//
// Roofline: ~1.6e11 FLOP total vs ~6 MB of HBM traffic -> pure compute.
// All three MLP layers + the 8 Jacobian-trace tangent matmuls run on
// V_WMMA_F32_16X16X32_F16 (300 static WMMAs per RHS body).
// Each wave owns 16 samples for the whole 8-step integration (samples are
// independent -> zero global synchronization).
//
// Round-4 change vs round 3: removed amdgpu_num_vgpr(256). The clamp caused
// scratch spilling (vmem 72 -> 1099, scratch_load in the WMMA inner loop),
// which is far worse than the s_set_vgpr_msb SALU toggles it eliminated.
// Occupancy is LDS-capped regardless, and the wide register file lets the
// scheduler keep two WMMA accumulator chains + fragment loads in flight.
// ---------------------------------------------------------------------------

#define T_PTS 9
#define NTOT  16384
#define DZ    8
#define HID   128
#define STRH  136   // f16 row stride for 128-wide arrays (272B -> 4-dword bank rotation)
#define STRX  40    // f16 row stride for 32-wide arrays  (80B  -> 20-dword bank rotation)

typedef _Float16 v16h __attribute__((ext_vector_type(16)));
typedef _Float16 v8h  __attribute__((ext_vector_type(8)));
typedef float    v8f  __attribute__((ext_vector_type(8)));

__device__ __forceinline__ float fast_tanh(float x){
  // tanh(x) = 1 - 2/(exp(2x)+1); v_exp_f32 + v_rcp_f32 (TRANS, co-exec w/ WMMA).
  x = fminf(10.f, fmaxf(-10.f, x));
  const float e = __expf(2.f * x);
  return fmaf(-2.f, __builtin_amdgcn_rcpf(e + 1.f), 1.f);
}

// A-fragment (16x32 f16), per CDNA5 ISA 7.12.2:
// lane L holds row M = L&15; element e holds K = kk*32 + (e/8)*16 + (L>>4)*8 + e%8.
__device__ __forceinline__ v16h load_fragA(const _Float16* row, int kk, int sub){
  const v8h lo = *(const v8h*)(row + kk*32 + sub*8);
  const v8h hi = *(const v8h*)(row + kk*32 + 16 + sub*8);
  v16h r;
#pragma unroll
  for (int i = 0; i < 8; ++i){ r[i] = lo[i]; r[8+i] = hi[i]; }
  return r;
}

// B-fragment (32x16 f16): lane L holds column N = L&15; element e holds
// K = kk*32 + (e/8)*16 + (L>>4)*8 + e%8; loaded from a row of the transposed weights Wt[N][K].
__device__ __forceinline__ v16h load_fragB(const _Float16* row, int kk, int sub){
  const v8h lo = *(const v8h*)(row + kk*32 + sub*8);
  const v8h hi = *(const v8h*)(row + kk*32 + 16 + sub*8);
  v16h r;
#pragma unroll
  for (int i = 0; i < 8; ++i){ r[i] = lo[i]; r[8+i] = hi[i]; }
  return r;
}

__global__ void __launch_bounds__(64)
cnf_rk4_wmma(const float* __restrict__ gts,
             const float* __restrict__ gz0,
             const float* __restrict__ glp0,
             const float* __restrict__ gW1, const float* __restrict__ gb1,
             const float* __restrict__ gW2, const float* __restrict__ gb2,
             const float* __restrict__ gW3, const float* __restrict__ gb3,
             float* __restrict__ out)
{
  // ---- LDS: 65,440 B (weights shared by both waves) ----
  __shared__ __align__(16) _Float16 sW2t [HID][STRH]; // W2^T [n][k]      (B: layer2+trace)
  __shared__ __align__(16) _Float16 sW3t [16][STRH];  // W3^T [j][k], j>=8 zero (B: layer3)
  __shared__ __align__(16) _Float16 sW1h [DZ][HID];   // W1 rows 0..7 (tangent seeds, bcast)
  __shared__ __align__(16) _Float16 sW1tB[HID][STRX]; // aug W1^T [n][k]: k<9 W1, k==9 b1, else 0
  __shared__ __align__(16) _Float16 sX   [2][16][STRX]; // per-wave x = [z, t, 1, 0...] (A: layer1)
  __shared__ __align__(16) _Float16 sH   [2][16][STRH]; // per-wave h1, recycled for h2
  __shared__ _Float16 sB2h[HID];
  __shared__ float    sB3[DZ];
  __shared__ float    sZ0[2][16][DZ], sKa[2][16][DZ];
  __shared__ float    sLp[2][16], sLa[2][16], sTr[2][16];

  const int tid  = threadIdx.x;
  const int lane = tid & 31;
  const int w    = tid >> 5;          // wave in block (0..1)
  const int sub  = lane >> 4;         // half-wave
  const int l16  = lane & 15;
  const int gbase = (blockIdx.x*2 + w)*16;

  float* outz  = out;                                // zt   [T,N,8]
  float* outlp = out + (size_t)T_PTS*NTOT*DZ;        // logp [T,N,1]

  // ---- stage weights into LDS (transposed f16 copies = WMMA B operands) ----
  for (int idx = tid; idx < HID*HID; idx += 64){
    const int k = idx >> 7, n = idx & 127;
    sW2t[n][k] = (_Float16)gW2[idx];                 // gW2 row-major [k][n]
  }
  for (int idx = tid; idx < 16*HID; idx += 64){
    const int j = idx >> 7, k = idx & 127;
    sW3t[j][k] = (j < DZ) ? (_Float16)gW3[k*DZ + j] : (_Float16)0.f;
  }
  for (int idx = tid; idx < DZ*HID; idx += 64){
    const int j = idx >> 7, n = idx & 127;
    sW1h[j][n] = (_Float16)gW1[j*HID + n];
  }
  for (int idx = tid; idx < HID*32; idx += 64){      // augmented W1 (bias via const-1 col)
    const int n = idx >> 5, k = idx & 31;
    float v = 0.f;
    if (k < DZ+1)       v = gW1[k*HID + n];
    else if (k == DZ+1) v = gb1[n];
    sW1tB[n][k] = (_Float16)v;
  }
  for (int n = tid; n < HID; n += 64) sB2h[n] = (_Float16)gb2[n];
  if (tid < DZ) sB3[tid] = gb3[tid];

  v16h hone;
#pragma unroll
  for (int i = 0; i < 16; ++i) hone[i] = (_Float16)1.0f;

  // ---- load state tile; emit t=0 outputs; build initial x-buffer ----
  {
    const int d0 = sub*4;
    const float4 v = *(const float4*)&gz0[(size_t)(gbase + l16)*DZ + d0];
    sZ0[w][l16][d0+0] = v.x; sZ0[w][l16][d0+1] = v.y;
    sZ0[w][l16][d0+2] = v.z; sZ0[w][l16][d0+3] = v.w;
    sX[w][l16][d0+0] = (_Float16)v.x; sX[w][l16][d0+1] = (_Float16)v.y;
    sX[w][l16][d0+2] = (_Float16)v.z; sX[w][l16][d0+3] = (_Float16)v.w;
    *(float4*)&outz[(size_t)(gbase + l16)*DZ + d0] = v;
    if (lane < 16){
      const float l = glp0[gbase + lane];
      sLp[w][lane] = l;
      outlp[gbase + lane] = l;
      sX[w][lane][DZ]   = (_Float16)gts[0];    // t column
      sX[w][lane][DZ+1] = (_Float16)1.0f;      // bias column
#pragma unroll
      for (int c = DZ+2; c < 32; ++c) sX[w][lane][c] = (_Float16)0.f;
    }
  }
  __syncthreads();

  for (int step = 0; step < T_PTS-1; ++step){
    const float t0 = gts[step], t1 = gts[step+1];
    const float dt = t1 - t0;

    { // zero RK4 accumulators
      const int d0 = sub*4;
#pragma unroll
      for (int i = 0; i < 4; ++i) sKa[w][l16][d0+i] = 0.f;
      if (lane < 16) sLa[w][lane] = 0.f;
    }
    __syncthreads();

    for (int st = 0; st < 4; ++st){
      const float wc  = (st==1 || st==2) ? 2.f : 1.f;  // quadrature weight
      const float acn = (st==2) ? 1.f : 0.5f;          // NEXT stage offset (unused st==3)

      // ---- layer 1: h1 = tanh([z,t,1,0..] @ W1aug), one WMMA per N-tile ----
      {
        const v16h ax = load_fragA(&sX[w][l16][0], 0, sub);
        const v8f zc = {};
#pragma unroll
        for (int nt = 0; nt < 8; ++nt){
          const int ncol = nt*16 + l16;
          const v8f acc = __builtin_amdgcn_wmma_f32_16x16x32_f16(
              false, ax, false, load_fragB(&sW1tB[ncol][0], 0, sub),
              (short)0, zc, false, false);
#pragma unroll
          for (int r = 0; r < 8; ++r)   // C layout: M = r + 8*sub, N = ncol
            sH[w][r + 8*sub][ncol] = (_Float16)fast_tanh(acc[r]);
        }
      }
      __syncthreads();

      // ---- layer 2: h2 = tanh(h1 @ W2 + b2), 8 N-tiles x 4 K-steps ----
      v16h a1[4], s1f[4];
      {
        const _Float16* hrow = &sH[w][l16][0];
#pragma unroll
        for (int kk = 0; kk < 4; ++kk) a1[kk] = load_fragA(hrow, kk, sub);
      }
#pragma unroll
      for (int kk = 0; kk < 4; ++kk) s1f[kk] = hone - a1[kk]*a1[kk];   // tanh' = 1-h1^2

#pragma unroll
      for (int nt = 0; nt < 8; ++nt){
        const int ncol = nt*16 + l16;
        const _Float16* wrow = &sW2t[ncol][0];
        v8f acc = {};
#pragma unroll
        for (int kk = 0; kk < 4; ++kk)
          acc = __builtin_amdgcn_wmma_f32_16x16x32_f16(
              false, a1[kk], false, load_fragB(wrow, kk, sub),
              (short)0, acc, false, false);
        const float bb = (float)sB2h[ncol];
#pragma unroll
        for (int r = 0; r < 8; ++r)   // recycle sH as h2 (a1 already register-resident)
          sH[w][r + 8*sub][ncol] = (_Float16)fast_tanh(acc[r] + bb);
      }
      __syncthreads();

      // ---- layer 3 + fused RK4 stage update ----
      // dz[M][j] lands in the owning lane; that lane RMWs sKa and writes the
      // NEXT stage's x-buffer (f16), eliminating sZt/sKc round-trips.
      {
        const _Float16* hrow = &sH[w][l16][0];
        const _Float16* wrow = &sW3t[l16][0];
        v8f acc = {};
#pragma unroll
        for (int kk = 0; kk < 4; ++kk)
          acc = __builtin_amdgcn_wmma_f32_16x16x32_f16(
              false, load_fragA(hrow, kk, sub), false, load_fragB(wrow, kk, sub),
              (short)0, acc, false, false);
        if (l16 < DZ){
#pragma unroll
          for (int r = 0; r < 8; ++r){
            const int M = r + 8*sub;
            const float kv = acc[r] + sB3[l16];
            sKa[w][M][l16] += wc * kv;
            if (st < 3)
              sX[w][M][l16] = (_Float16)(sZ0[w][M][l16] + acn*dt*kv);
          }
        } else if (l16 == DZ && st < 3){
          const _Float16 tn = (_Float16)(t0 + acn*dt);
#pragma unroll
          for (int r = 0; r < 8; ++r) sX[w][r + 8*sub][DZ] = tn;
        }
      }

      // ---- exact Jacobian trace: Dh1_j = W1[j,:] .* (1-h1^2),
      //      Dh2pre_j = Dh1_j @ W2 (WMMA), tr += Dh2pre_j .* (1-h2^2) . W3[:,j]
      //      N-tile outer: W2 B-fragments reused across all 8 directions;
      //      W1-row fragment loads are same-address LDS broadcasts.
      float tracc[8] = {0.f,0.f,0.f,0.f,0.f,0.f,0.f,0.f};
#pragma unroll
      for (int nt = 0; nt < 8; ++nt){
        const int ncol = nt*16 + l16;
        const _Float16* wrow = &sW2t[ncol][0];
        v16h bf[4];
#pragma unroll
        for (int kk = 0; kk < 4; ++kk) bf[kk] = load_fragB(wrow, kk, sub);
        float s2r[8];
#pragma unroll
        for (int r = 0; r < 8; ++r){
          const float hh = (float)sH[w][r + 8*sub][ncol];
          s2r[r] = 1.f - hh*hh;                       // tanh' at layer 2, C-layout aligned
        }
#pragma unroll
        for (int j = 0; j < DZ; ++j){
          const float w3v = (float)sW3t[j][ncol];     // W3[ncol][j]
          const _Float16* w1row = &sW1h[j][0];
          v8f acc = {};
#pragma unroll
          for (int kk = 0; kk < 4; ++kk){
            const v16h af = s1f[kk] * load_fragA(w1row, kk, sub);  // packed-f16 VALU
            acc = __builtin_amdgcn_wmma_f32_16x16x32_f16(
                false, af, false, bf[kk], (short)0, acc, false, false);
          }
#pragma unroll
          for (int r = 0; r < 8; ++r) tracc[r] = fmaf(acc[r]*s2r[r], w3v, tracc[r]);
        }
      }
      // reduce over the 16 lanes of each half (samples 0-7 in lanes 0-15, 8-15 in 16-31)
#pragma unroll
      for (int r = 0; r < 8; ++r){
        float v = tracc[r];
        for (int m = 1; m < 16; m <<= 1) v += __shfl_xor(v, m, 16);
        if (l16 == 0) sTr[w][r + 8*sub] = v;
      }
      __syncthreads();

      if (lane < 16) sLa[w][lane] -= wc * sTr[w][lane];   // dlogp/dt = -tr
    } // stages

    // ---- combine, emit outputs for t_{step+1}, seed next step's x-buffer ----
    {
      const int d0 = sub*4;
      const float h6 = dt * (1.f/6.f);
      float4 v;
      v.x = (sZ0[w][l16][d0+0] += h6*sKa[w][l16][d0+0]);
      v.y = (sZ0[w][l16][d0+1] += h6*sKa[w][l16][d0+1]);
      v.z = (sZ0[w][l16][d0+2] += h6*sKa[w][l16][d0+2]);
      v.w = (sZ0[w][l16][d0+3] += h6*sKa[w][l16][d0+3]);
      *(float4*)&outz[(size_t)(step+1)*NTOT*DZ + (size_t)(gbase + l16)*DZ + d0] = v;
      sX[w][l16][d0+0] = (_Float16)v.x; sX[w][l16][d0+1] = (_Float16)v.y;
      sX[w][l16][d0+2] = (_Float16)v.z; sX[w][l16][d0+3] = (_Float16)v.w;
      if (lane < 16){
        const float l = (sLp[w][lane] += h6*sLa[w][lane]);
        outlp[(size_t)(step+1)*NTOT + gbase + lane] = l;
        sX[w][lane][DZ] = (_Float16)t1;    // t0 of next step
      }
    }
    __syncthreads();
  } // steps
}

extern "C" void kernel_launch(void* const* d_in, const int* in_sizes, int n_in,
                              void* d_out, int out_size, void* d_ws, size_t ws_size,
                              hipStream_t stream) {
  (void)in_sizes; (void)n_in; (void)out_size; (void)d_ws; (void)ws_size;
  const float* gts  = (const float*)d_in[0];   // ts   [9]
  const float* gz0  = (const float*)d_in[1];   // z0   [16384,8]
  const float* glp0 = (const float*)d_in[2];   // logp [16384,1]
  const float* gW1  = (const float*)d_in[3];   // [9,128]
  const float* gb1  = (const float*)d_in[4];   // [128]
  const float* gW2  = (const float*)d_in[5];   // [128,128]
  const float* gb2  = (const float*)d_in[6];   // [128]
  const float* gW3  = (const float*)d_in[7];   // [128,8]
  const float* gb3  = (const float*)d_in[8];   // [8]

  dim3 grid(NTOT / 32);   // 512 blocks, 2 waves/block, 16 samples/wave
  dim3 block(64);
  cnf_rk4_wmma<<<grid, block, 0, stream>>>(gts, gz0, glp0, gW1, gb1, gW2, gb2,
                                           gW3, gb3, (float*)d_out);
}
